// CRF_24172075941977
// MI455X (gfx1250) — matrix operand order repaired and checked by reading, not compiled, not run on hardware
//
#include <hip/hip_runtime.h>
#include <hip/hip_bf16.h>

// ---------------------------------------------------------------------------
// CRF mean-field inference, MI455X (gfx1250, wave32, WMMA).
//
// Strategy:
//   G = Kb + Kg  (N x N, N=9216) precomputed once in fp16, stored directly in
//   the V_WMMA_F32_16X16X32_F16 A-fragment swizzled layout (16x32 tiles).
//   Each iteration: comb = G @ probs_pad16 via v_wmma (f32 accum), then the
//   Potts 3x3 conv collapses to upd[c] = box3(sum_c comb) - box3(comb[c]),
//   then softmax(input - 3*upd).
// ---------------------------------------------------------------------------

typedef __attribute__((ext_vector_type(16))) _Float16 v16h;
typedef __attribute__((ext_vector_type(8)))  _Float16 v8h;
typedef __attribute__((ext_vector_type(8)))  float    v8f;

#define HH    96
#define WW    96
#define NPIX  9216          // 96*96
#define NCH   5
#define KTILES 288          // 9216 / 32  (K tiles of the big GEMM)
#define RTILES 576          // 9216 / 16  (output row tiles)

// workspace offsets (bytes); total ~170.5 MB
#define OFF_G    0ull
#define SZ_G     (9216ull * 9216ull * 2ull)          // fp16 G, swizzled tiles
#define OFF_B    (OFF_G + SZ_G)
#define SZ_B     (9216ull * 16ull * 2ull)            // fp16 B, swizzled tiles
#define OFF_COMB (OFF_B + SZ_B)
#define SZ_COMB  (5ull * 9216ull * 4ull)             // f32 comb [c][n]
#define OFF_PROB (OFF_COMB + SZ_COMB)                // f32 probs [c][n]

// CRF hyperparameters
#define INV_SS2   (1.0f / 25.0f)    // 1/sig_spatial^2   (both kernels: 5.0)
#define INV_SC2   (4.0f)            // 1/sig_color^2     (0.5)
#define UPD_FAC   (3.0f)

// ---------------------------------------------------------------------------
// B-fragment pack helper: B is 32x16 (K x N) fp16.  Per ISA 7.12.2:
// lane = N + 16*((K%16)/8); element i (=2*vgpr+half) = 8*(K/16) + (K%8).
// Stored tiled: tile kt = m/32 holds 512 halves at kt*512 + lane*16 + i.
__device__ __forceinline__ void pack_B(_Float16* Bswz, int m, const float* p) {
    int kt = m >> 5, Km = m & 31;
    int g  = (Km >> 3) & 1, i = ((Km >> 4) << 3) | (Km & 7);
    _Float16* base = Bswz + (size_t)kt * 512 + i;
#pragma unroll
    for (int c = 0; c < NCH; ++c)
        base[(size_t)(c + 16 * g) * 16] = (_Float16)p[c];
}

// ---------------------------------------------------------------------------
// Zero the swizzled B buffer once (channels 5..15 must stay zero forever).
__global__ void k_zero_b(unsigned int* b) {
    int i = blockIdx.x * blockDim.x + threadIdx.x;   // 73728 uints exactly
    b[i] = 0u;
}

// ---------------------------------------------------------------------------
// Build G = exp(-0.5*d2_bilateral) + exp(-0.5*d2_gaussian) in fp16,
// stored in A-fragment swizzled layout.  A is 16x32 (M x K); per ISA 7.12.2:
// lane = M + 16*g with g = (K%16)/8 ; element i = 8*(K/16) + (K%8).
// Each thread produces 8 consecutive K (aligned) => one contiguous v8h store.
__global__ void k_build_g(const float* __restrict__ ref, _Float16* __restrict__ G) {
    int gid = blockIdx.x * blockDim.x + threadIdx.x;  // 9216*1152 threads
    int n   = gid / 1152;
    int m0  = (gid - n * 1152) * 8;

    int   yn = n / WW, xn = n - yn * WW;
    float rn = ref[n], gn = ref[NPIX + n], bn = ref[2 * NPIX + n];

    v8h vals;
#pragma unroll
    for (int t = 0; t < 8; ++t) {
        int   m  = m0 + t;
        int   ym = m / WW, xm = m - ym * WW;
        float dy = (float)(yn - ym), dx = (float)(xn - xm);
        float ds = (dy * dy + dx * dx) * INV_SS2;              // spatial term
        float dr = rn - ref[m];
        float dg = gn - ref[NPIX + m];
        float db = bn - ref[2 * NPIX + m];
        float dc = (dr * dr + dg * dg + db * db) * INV_SC2;    // color term
        float v  = __expf(-0.5f * (ds + dc)) + __expf(-0.5f * ds);
        vals[t]  = (_Float16)v;
    }

    int M = n & 15;
    int K0 = m0 & 31;
    int g  = (K0 >> 3) & 1;
    int i0 = (K0 >> 4) << 3;                  // 8*(K0/16), K0%8 == 0
    int lane = M + 16 * g;
    size_t tile = (size_t)(n >> 4) * KTILES + (size_t)(m0 >> 5);
    *(v8h*)(G + tile * 512 + (size_t)lane * 16 + i0) = vals;
}

// ---------------------------------------------------------------------------
// Initial softmax over channels; writes probs (f32) and packs swizzled B.
__global__ void k_softmax_init(const float* __restrict__ logits,
                               float* __restrict__ probs,
                               _Float16* __restrict__ Bswz) {
    int n = blockIdx.x * blockDim.x + threadIdx.x;   // 9216 threads exactly
    float v[NCH], p[NCH];
    float mx = -3.0e38f;
#pragma unroll
    for (int c = 0; c < NCH; ++c) { v[c] = logits[c * NPIX + n]; mx = fmaxf(mx, v[c]); }
    float s = 0.f;
#pragma unroll
    for (int c = 0; c < NCH; ++c) { p[c] = __expf(v[c] - mx); s += p[c]; }
    float inv = 1.0f / s;
#pragma unroll
    for (int c = 0; c < NCH; ++c) { p[c] *= inv; probs[c * NPIX + n] = p[c]; }
    pack_B(Bswz, n, p);
}

// ---------------------------------------------------------------------------
// comb[16 rows x 5 cols per tile] = G(tile row) @ B.  One block per 16-row
// tile; 8 waves split the 288 K-tiles (36 WMMAs each), LDS reduction.
__global__ __launch_bounds__(256) void k_gemm(const _Float16* __restrict__ G,
                                              const _Float16* __restrict__ Bswz,
                                              float* __restrict__ comb) {
    __shared__ float red[8 * 256];
    int tid = threadIdx.x;
    int w = tid >> 5, lane = tid & 31;
    int rowTile = blockIdx.x;

    const v16h* Abase = (const v16h*)(G + (size_t)rowTile * KTILES * 512);
    const v16h* Bbase = (const v16h*)Bswz;

    v8f acc = {};
    int kt = w * (KTILES / 8);
#pragma unroll 4
    for (int kk = 0; kk < KTILES / 8; ++kk, ++kt) {
        v16h a = Abase[kt * 32 + lane];   // fully coalesced 32B/lane
        v16h b = Bbase[kt * 32 + lane];
        acc = __builtin_amdgcn_wmma_f32_16x16x32_f16(
                  false, a, false, b, (short)0, acc, false, false);
    }

#pragma unroll
    for (int j = 0; j < 8; ++j) red[w * 256 + lane * 8 + j] = acc[j];
    __syncthreads();

    // reduce 8 partial 16x16 f32 tiles; D layout: lane%16 = col(N=channel),
    // vgpr j -> row M = j + 8*(lane/16).
    int lane2 = tid >> 3, j = tid & 7;
    float s = 0.f;
#pragma unroll
    for (int w2 = 0; w2 < 8; ++w2) s += red[w2 * 256 + lane2 * 8 + j];
    int c = lane2 & 15, g = lane2 >> 4;
    int row = rowTile * 16 + j + 8 * g;
    if (c < NCH) comb[c * NPIX + row] = s;
}

// ---------------------------------------------------------------------------
// Potts update: upd[c] = box3(sum_c comb) - box3(comb[c]) (edge-clamped),
// then softmax(input - 3*upd).  Writes probs (to ws or d_out) + packs B.
__global__ void k_update(const float* __restrict__ logits,
                         const float* __restrict__ comb,
                         float* __restrict__ dst,
                         _Float16* __restrict__ Bswz) {
    int n = blockIdx.x * blockDim.x + threadIdx.x;   // 9216 threads exactly
    int y = n / WW, x = n - y * WW;

    float s[NCH] = {0.f, 0.f, 0.f, 0.f, 0.f};
#pragma unroll
    for (int dy = -1; dy <= 1; ++dy) {
        int yy = min(max(y + dy, 0), HH - 1);
#pragma unroll
        for (int dx = -1; dx <= 1; ++dx) {
            int xx = min(max(x + dx, 0), WW - 1);
            int q = yy * WW + xx;
#pragma unroll
            for (int c = 0; c < NCH; ++c) s[c] += comb[c * NPIX + q];
        }
    }
    float tot = 0.f;
#pragma unroll
    for (int c = 0; c < NCH; ++c) tot += s[c];

    float v[NCH], p[NCH];
    float mx = -3.0e38f;
#pragma unroll
    for (int c = 0; c < NCH; ++c) {
        v[c] = logits[c * NPIX + n] - UPD_FAC * (tot - s[c]);
        mx = fmaxf(mx, v[c]);
    }
    float sm = 0.f;
#pragma unroll
    for (int c = 0; c < NCH; ++c) { p[c] = __expf(v[c] - mx); sm += p[c]; }
    float inv = 1.0f / sm;
#pragma unroll
    for (int c = 0; c < NCH; ++c) { p[c] *= inv; dst[c * NPIX + n] = p[c]; }
    pack_B(Bswz, n, p);
}

// ---------------------------------------------------------------------------
extern "C" void kernel_launch(void* const* d_in, const int* in_sizes, int n_in,
                              void* d_out, int out_size, void* d_ws, size_t ws_size,
                              hipStream_t stream) {
    const float* input = (const float*)d_in[0];   // [1,5,96,96] f32
    const float* refc  = (const float*)d_in[1];   // [1,3,96,96] f32
    float* out = (float*)d_out;                   // [1,5,96,96] f32

    char* ws = (char*)d_ws;                       // needs ~170.5 MB
    _Float16* G    = (_Float16*)(ws + OFF_G);
    _Float16* Bswz = (_Float16*)(ws + OFF_B);
    float*    comb = (float*)(ws + OFF_COMB);
    float*    prob = (float*)(ws + OFF_PROB);

    // zero B padding channels 5..15 (stays zero across iterations)
    k_zero_b<<<288, 256, 0, stream>>>((unsigned int*)Bswz);

    // one-time fp16 kernel matrix, swizzled for WMMA A fragments
    k_build_g<<<(NPIX * 1152) / 256, 256, 0, stream>>>(refc, G);

    // Q0 = softmax(unaries)
    k_softmax_init<<<NPIX / 256, 256, 0, stream>>>(input, prob, Bswz);

    for (int it = 0; it < 5; ++it) {
        k_gemm<<<RTILES, 256, 0, stream>>>(G, Bswz, comb);
        float* dst = (it == 4) ? out : prob;
        k_update<<<NPIX / 256, 256, 0, stream>>>(input, comb, dst, Bswz);
    }
}
